// FeatureProjector_13065290515076
// MI455X (gfx1250) — matrix operand order repaired and checked by reading, not compiled, not run on hardware
//
#include <hip/hip_runtime.h>
#include <math.h>

typedef __attribute__((ext_vector_type(16))) _Float16 v16h;
typedef __attribute__((ext_vector_type(8)))  _Float16 v8h;
typedef __attribute__((ext_vector_type(8)))  float    v8f;

#define D_OUT 192

// ---------------------------------------------------------------------------
// Prep: Wt[c][k] = f16(W[k][c]) (zero-padded to kpad), gW[c] = sum_k gamma[k]W[k][c],
// bW[c] = sum_k beta[k]W[k][c].  One block per output column c.
// ---------------------------------------------------------------------------
__global__ __launch_bounds__(128)
void prep_weights(const float* __restrict__ W, const float* __restrict__ gamma,
                  const float* __restrict__ beta, _Float16* __restrict__ Wt,
                  float* __restrict__ gW, float* __restrict__ bW,
                  int deff, int kpad)
{
    const int c = blockIdx.x;                 // 0..191
    float sg = 0.f, sb = 0.f;
    for (int k = threadIdx.x; k < kpad; k += 128) {
        float w = 0.f;
        if (k < deff) {
            w = W[(size_t)k * D_OUT + c];
            sg += gamma[k] * w;
            sb += beta[k]  * w;
        }
        Wt[(size_t)c * kpad + k] = (_Float16)w;
    }
    __shared__ float s0[128], s1[128];
    s0[threadIdx.x] = sg; s1[threadIdx.x] = sb;
    __syncthreads();
    for (int off = 64; off > 0; off >>= 1) {
        if (threadIdx.x < off) {
            s0[threadIdx.x] += s0[threadIdx.x + off];
            s1[threadIdx.x] += s1[threadIdx.x + off];
        }
        __syncthreads();
    }
    if (threadIdx.x == 0) { gW[c] = s0[0]; bW[c] = s1[0]; }
}

// ---------------------------------------------------------------------------
// Main fused branch kernel: 32 rows per block, 128 threads (4 waves).
// Stage 1: stream x, build LN stats (f32), store f16(x*gamma) to LDS.
// Stage 2: WMMA f16 GEMM (K padded), epilogue folds LN scalars + GELU.
// ---------------------------------------------------------------------------
template<int D_IN, int KPAD, bool TIME>
__global__ __launch_bounds__(128)
void branch_kernel(const float* __restrict__ x, const float* __restrict__ ts,
                   const float* __restrict__ gamma,
                   const _Float16* __restrict__ Wt,
                   const float* __restrict__ gW, const float* __restrict__ bW,
                   const float* __restrict__ w0p, const float* __restrict__ wp,
                   const float* __restrict__ bp,
                   float* __restrict__ out, int nrows)
{
    constexpr int DEFF = D_IN + (TIME ? 12 : 0);
    constexpr int STR  = KPAD + 8;                 // padded LDS row stride (halves)
    __shared__ __attribute__((aligned(32))) _Float16 Ash[32 * STR];
    __shared__ float rowRs[32], rowMuRs[32];

    const int tid  = threadIdx.x;
    const int row0 = blockIdx.x * 32;

    // -------- Stage 1: load + stats + f16(x*gamma) into LDS --------
    {
        const int r  = tid >> 2;                   // 0..31 (4 threads per row)
        const int l4 = tid & 3;
        const long grow = (long)row0 + r;
        const bool inb  = grow < (long)nrows;
        const float* xr = x + (size_t)grow * D_IN;
        _Float16* arow  = &Ash[r * STR];

        float s = 0.f, ss = 0.f;
        for (int c = l4 * 4; c < D_IN; c += 16) {
            float4 v = inb ? *(const float4*)(xr + c) : make_float4(0.f, 0.f, 0.f, 0.f);
            const float4 g = *(const float4*)(gamma + c);
            s  += v.x + v.y + v.z + v.w;
            ss += v.x * v.x + v.y * v.y + v.z * v.z + v.w * v.w;
            arow[c + 0] = (_Float16)(v.x * g.x);
            arow[c + 1] = (_Float16)(v.y * g.y);
            arow[c + 2] = (_Float16)(v.z * g.z);
            arow[c + 3] = (_Float16)(v.w * g.w);
        }
        if (TIME) {
            if (l4 == 0) {
                const float t = inb ? ts[grow] : 0.f;
                float v0 = w0p[0] * t + bp[0];
                if (!inb) v0 = 0.f;
                s += v0; ss += v0 * v0;
                arow[D_IN] = (_Float16)(v0 * gamma[D_IN]);
                #pragma unroll
                for (int j = 0; j < 11; ++j) {
                    float vj = sinf(wp[j] * t + bp[1 + j]);
                    if (!inb) vj = 0.f;
                    s += vj; ss += vj * vj;
                    arow[D_IN + 1 + j] = (_Float16)(vj * gamma[D_IN + 1 + j]);
                }
            }
            for (int c = DEFF + l4; c < KPAD; c += 4)   // zero K padding
                arow[c] = (_Float16)0.f;
        }
        // reduce across the 4 lanes of this row (wave32 shuffles)
        s  += __shfl_xor(s, 1);  s  += __shfl_xor(s, 2);
        ss += __shfl_xor(ss, 1); ss += __shfl_xor(ss, 2);
        if (l4 == 0) {
            const float inv = 1.0f / (float)DEFF;
            const float mu  = s * inv;
            const float var = ss * inv - mu * mu;        // biased variance
            const float rs  = rsqrtf(var + 1e-5f);
            rowRs[r]   = rs;
            rowMuRs[r] = mu * rs;
        }
    }
    __syncthreads();

    // -------- Stage 2: WMMA f16 GEMM, f32 accumulate --------
    const int lane = tid & 31;
    const int wv   = tid >> 5;         // wave -> column tiles wv, wv+4, wv+8
    const int l16  = lane & 15;
    const int hs   = lane >> 4;        // half-wave select

    v8f acc[2][3] = {};

    for (int k0 = 0; k0 < KPAD; k0 += 32) {
        // B fragments: lane<16 holds K=k0..k0+15 of column c; lane>=16 holds K=k0+16..k0+31
        v16h bf[3];
        #pragma unroll
        for (int j = 0; j < 3; ++j) {
            const int col = (wv + 4 * j) * 16 + l16;
            bf[j] = *(const v16h*)(Wt + (size_t)col * KPAD + k0 + hs * 16);
        }
        #pragma unroll
        for (int m = 0; m < 2; ++m) {
            // A fragment: lane<16 row=l16 K=k0..k0+15; lane>=16 K=k0+16..k0+31
            const _Float16* ap = &Ash[(m * 16 + l16) * STR + k0 + hs * 16];
            const v8h alo = *(const v8h*)ap;
            const v8h ahi = *(const v8h*)(ap + 8);
            v16h af;
            #pragma unroll
            for (int i = 0; i < 8; ++i) { af[i] = alo[i]; af[i + 8] = ahi[i]; }
            #pragma unroll
            for (int j = 0; j < 3; ++j)
                acc[m][j] = __builtin_amdgcn_wmma_f32_16x16x32_f16(
                    false, af, false, bf[j], (short)0, acc[m][j], false, false);
        }
    }

    // -------- Epilogue: fold LN scalars, exact GELU, store --------
    #pragma unroll
    for (int j = 0; j < 3; ++j) {
        const int col = (wv + 4 * j) * 16 + l16;
        const float gw = gW[col];
        const float bw = bW[col];
        #pragma unroll
        for (int m = 0; m < 2; ++m) {
            #pragma unroll
            for (int i = 0; i < 8; ++i) {
                const int  rl   = m * 16 + hs * 8 + i;   // C/D layout: VGPR i, half-wave hs
                const long grow = (long)row0 + rl;
                if (grow < (long)nrows) {
                    const float y  = rowRs[rl] * acc[m][j][i] - rowMuRs[rl] * gw + bw;
                    const float ge = 0.5f * y * (1.0f + erff(y * 0.70710678118654752f));
                    out[grow * (long)D_OUT + col] = ge;
                }
            }
        }
    }
}

// ---------------------------------------------------------------------------
extern "C" void kernel_launch(void* const* d_in, const int* in_sizes, int n_in,
                              void* d_out, int out_size, void* d_ws, size_t ws_size,
                              hipStream_t stream)
{
    const float* x_bill  = (const float*)d_in[0];
    const float* ts_bill = (const float*)d_in[1];
    const float* x_bv    = (const float*)d_in[2];
    const float* ts_bv   = (const float*)d_in[3];
    const float* x_leg   = (const float*)d_in[4];
    const float* x_com   = (const float*)d_in[5];
    const float* w0      = (const float*)d_in[6];
    const float* w       = (const float*)d_in[7];
    const float* b       = (const float*)d_in[8];
    const float* g_bill  = (const float*)d_in[9];
    const float* be_bill = (const float*)d_in[10];
    const float* W_bill  = (const float*)d_in[11];
    const float* g_bv    = (const float*)d_in[12];
    const float* be_bv   = (const float*)d_in[13];
    const float* W_bv    = (const float*)d_in[14];
    const float* g_leg   = (const float*)d_in[15];
    const float* be_leg  = (const float*)d_in[16];
    const float* W_leg   = (const float*)d_in[17];
    const float* g_com   = (const float*)d_in[18];
    const float* be_com  = (const float*)d_in[19];
    const float* W_com   = (const float*)d_in[20];

    const int Nr = in_sizes[1];                  // ts_bill length == N
    float* out = (float*)d_out;

    constexpr int KP_BILL = 416, KP_BV = 800, KP_LEG = 256, KP_COM = 128;

    // workspace layout (f16 weight panels, then f32 gW/bW vectors)
    _Float16* WtB = (_Float16*)d_ws;
    _Float16* WtV = WtB + (size_t)KP_BILL * D_OUT;
    _Float16* WtL = WtV + (size_t)KP_BV   * D_OUT;
    _Float16* WtC = WtL + (size_t)KP_LEG  * D_OUT;
    float* fb  = (float*)(WtC + (size_t)KP_COM * D_OUT);
    float* gWb = fb;          float* bWb = fb + 192;
    float* gWv = fb + 384;    float* bWv = fb + 576;
    float* gWl = fb + 768;    float* bWl = fb + 960;
    float* gWc = fb + 1152;   float* bWc = fb + 1344;

    prep_weights<<<D_OUT, 128, 0, stream>>>(W_bill, g_bill, be_bill, WtB, gWb, bWb, 396, KP_BILL);
    prep_weights<<<D_OUT, 128, 0, stream>>>(W_bv,   g_bv,   be_bv,   WtV, gWv, bWv, 780, KP_BV);
    prep_weights<<<D_OUT, 128, 0, stream>>>(W_leg,  g_leg,  be_leg,  WtL, gWl, bWl, 256, KP_LEG);
    prep_weights<<<D_OUT, 128, 0, stream>>>(W_com,  g_com,  be_com,  WtC, gWc, bWc, 128, KP_COM);

    const int nb = (Nr + 31) / 32;
    branch_kernel<384, KP_BILL, true ><<<nb, 128, 0, stream>>>(
        x_bill, ts_bill, g_bill, WtB, gWb, bWb, w0, w, b, out, Nr);
    branch_kernel<768, KP_BV,   true ><<<nb, 128, 0, stream>>>(
        x_bv,   ts_bv,   g_bv,   WtV, gWv, bWv, w0, w, b, out + (size_t)Nr * D_OUT, Nr);
    branch_kernel<256, KP_LEG,  false><<<nb, 128, 0, stream>>>(
        x_leg,  nullptr, g_leg,  WtL, gWl, bWl, w0, w, b, out + (size_t)2 * Nr * D_OUT, Nr);
    branch_kernel<128, KP_COM,  false><<<nb, 128, 0, stream>>>(
        x_com,  nullptr, g_com,  WtC, gWc, bWc, w0, w, b, out + (size_t)3 * Nr * D_OUT, Nr);
}
